// MultiTaskNet_72464688218832
// MI455X (gfx1250) — matrix operand order repaired and checked by reading, not compiled, not run on hardware
//
#include <hip/hip_runtime.h>
#include <hip/hip_bf16.h>

typedef __attribute__((ext_vector_type(16))) _Float16 v16h;
typedef __attribute__((ext_vector_type(8)))  float    v8f;

#define EMB_DIM 32
#define H1 96
#define H2 64

__device__ __forceinline__ float4 ld4(const float* p) {
    return *reinterpret_cast<const float4*>(p);
}
__device__ __forceinline__ void st4(float* p, float4 v) {
    *reinterpret_cast<float4*>(p) = v;
}

// D = A(16x32 f16) x B(32x16 f16) + C(16x16 f32)
__device__ __forceinline__ v8f wmma_f16(v16h a, v16h b, v8f c) {
    return __builtin_amdgcn_wmma_f32_16x16x32_f16(
        /*neg_a=*/false, a, /*neg_b=*/false, b,
        /*c_mod=*/(short)0, c, /*reuse_a=*/false, /*reuse_b=*/false);
}

__global__ __launch_bounds__(256)
void multitask_fused_kernel(const int* __restrict__ user_ids,
                            const int* __restrict__ item_ids,
                            const float* __restrict__ U,
                            const float* __restrict__ Q,
                            const float* __restrict__ A,
                            const float* __restrict__ Bb,
                            const float* __restrict__ W1,
                            const float* __restrict__ b1,
                            const float* __restrict__ W2,
                            const float* __restrict__ b2,
                            float* __restrict__ out,
                            int batch) {
    const int lane  = threadIdx.x & 31;
    const int wave  = threadIdx.x >> 5;
    const int half  = lane >> 4;   // K-half selector (A/B frags), M-half (C)
    const int lm    = lane & 15;   // M for A rows, N for B/C columns
    const int wavesPerBlock = blockDim.x >> 5;
    const int globalWave = blockIdx.x * wavesPerBlock + wave;
    const int totalWaves = gridDim.x * wavesPerBlock;
    const int numTiles = batch >> 4;   // 16 rows per tile

    // ---- Resident W1^T fragments: Bfrag[kstep s][ntile t] ----
    // B-matrix (32x16, K x N) layout: lane l holds N = l%16,
// K = 32*s + {8*half..8*half+7} U {16+8*half..16+8*half+7}, packed 2 per VGPR.
    // B[k][n] = W1[n][k], W1 row-major [64 x 96].
    v16h Bfrag[3][4];
    #pragma unroll
    for (int s = 0; s < 3; ++s) {
        #pragma unroll
        for (int t = 0; t < 4; ++t) {
            const float* wp = W1 + (16 * t + lm) * H1 + 32 * s + 8 * half;
            float wr[16];
            st4(&wr[0],  ld4(wp + 0));
            st4(&wr[4],  ld4(wp + 4));
            st4(&wr[8],  ld4(wp + 16));
            st4(&wr[12], ld4(wp + 20));
            #pragma unroll
            for (int j = 0; j < 16; ++j)
                Bfrag[s][t][j] = (_Float16)wr[j];
        }
    }

    // Per-lane layer-1 bias and layer-2 weight for this lane's N columns.
    float b1v[4], w2v[4];
    #pragma unroll
    for (int t = 0; t < 4; ++t) {
        b1v[t] = b1[16 * t + lm];
        w2v[t] = W2[16 * t + lm];
    }
    const float b2v = b2[0];

    for (int tile = globalWave; tile < numTiles; tile += totalWaves) {
        const int rowBase = tile << 4;
        const int row = rowBase + lm;               // this lane's M row
        const int rc  = row < batch ? row : batch - 1;
        const int uid = user_ids[rc];
        const int iid = item_ids[rc];

        // ---- Gather the 16 K-values of this lane's row for U and Q ----
        // A-matrix 16x32 f16 layout: lane l holds M = l%16,
        // K in {8h..8h+7} U {16+8h..16+8h+7}, packed 2 per VGPR.
        const float* up = U + (long long)uid * EMB_DIM + 8 * half;
        const float* qp = Q + (long long)iid * EMB_DIM + 8 * half;
        float us[16], qs[16];
        st4(&us[0],  ld4(up + 0));  st4(&us[4],  ld4(up + 4));
        st4(&us[8],  ld4(up + 16)); st4(&us[12], ld4(up + 20));
        st4(&qs[0],  ld4(qp + 0));  st4(&qs[4],  ld4(qp + 4));
        st4(&qs[8],  ld4(qp + 16)); st4(&qs[12], ld4(qp + 20));

        v16h aU, aQ, aP;
        float dotp = 0.0f;
        #pragma unroll
        for (int j = 0; j < 16; ++j) {
            const float uu = us[j];
            const float qq = qs[j];
            aU[j] = (_Float16)uu;
            aQ[j] = (_Float16)qq;
            aP[j] = (_Float16)(uu * qq);
            dotp += uu * qq;
        }

        // ---- Layer 1: h = relu(mlp_in @ W1^T + b1), via 12 WMMAs ----
        v8f acc[4];
        #pragma unroll
        for (int t = 0; t < 4; ++t) {
            const float bv = b1v[t];
            acc[t] = (v8f){bv, bv, bv, bv, bv, bv, bv, bv};
        }
        #pragma unroll
        for (int t = 0; t < 4; ++t) acc[t] = wmma_f16(aU, Bfrag[0][t], acc[t]);
        #pragma unroll
        for (int t = 0; t < 4; ++t) acc[t] = wmma_f16(aQ, Bfrag[1][t], acc[t]);
        #pragma unroll
        for (int t = 0; t < 4; ++t) acc[t] = wmma_f16(aP, Bfrag[2][t], acc[t]);

        // ---- Layer 2: score[m] = sum_n relu(h[m,n]) * W2[n] + b2 ----
        // C layout: VGPR r -> M = r + 8*half, lane -> N = 16*t + lm.
        float p[8];
        #pragma unroll
        for (int r = 0; r < 8; ++r) {
            float s = 0.0f;
            #pragma unroll
            for (int t = 0; t < 4; ++t) {
                float hv = acc[t][r];
                hv = hv > 0.0f ? hv : 0.0f;
                s += hv * w2v[t];
            }
            // reduce the 16 N-lanes within this half
            s += __shfl_xor(s, 1, 32);
            s += __shfl_xor(s, 2, 32);
            s += __shfl_xor(s, 4, 32);
            s += __shfl_xor(s, 8, 32);
            p[r] = s + b2v;
        }
        // all 16 lanes of a half hold p[0..7] for rows rowBase+8*half+r
        if (lm == 0) {
            const int mb = rowBase + 8 * half;
            if (mb + 8 <= batch) {
                st4(out + batch + mb,     (float4){p[0], p[1], p[2], p[3]});
                st4(out + batch + mb + 4, (float4){p[4], p[5], p[6], p[7]});
            } else {
                for (int r = 0; r < 8; ++r)
                    if (mb + r < batch) out[batch + mb + r] = p[r];
            }
        }

        // ---- predictions[m] = dot(U_emb, Q_emb) + A[uid] + B[iid] ----
        const float dfull = dotp + __shfl_xor(dotp, 16, 32);
        if (half == 0 && row < batch) {
            out[row] = dfull + A[uid] + Bb[iid];
        }
    }
}

extern "C" void kernel_launch(void* const* d_in, const int* in_sizes, int n_in,
                              void* d_out, int out_size, void* d_ws, size_t ws_size,
                              hipStream_t stream) {
    const int*   user_ids = (const int*)  d_in[0];
    const int*   item_ids = (const int*)  d_in[1];
    const float* U        = (const float*)d_in[2];
    const float* Q        = (const float*)d_in[3];
    const float* A        = (const float*)d_in[4];
    const float* Bb       = (const float*)d_in[5];
    const float* W1       = (const float*)d_in[6];
    const float* b1       = (const float*)d_in[7];
    const float* W2       = (const float*)d_in[8];
    const float* b2       = (const float*)d_in[9];
    float* out = (float*)d_out;

    const int batch = in_sizes[0];
    const int threads = 256;                 // 8 wave32s per block
    const int wavesPerBlock = threads / 32;
    const int numTiles = (batch + 15) >> 4;
    int blocks = (numTiles + wavesPerBlock - 1) / wavesPerBlock;
    if (blocks > 4096) blocks = 4096;        // grid-stride over tiles
    if (blocks < 1) blocks = 1;

    multitask_fused_kernel<<<blocks, threads, 0, stream>>>(
        user_ids, item_ids, U, Q, A, Bb, W1, b1, W2, b2, out, batch);
}
